// GraphSAGENetwork_76046690943378
// MI455X (gfx1250) — matrix operand order repaired and checked by reading, not compiled, not run on hardware
//
#include <hip/hip_runtime.h>
#include <hip/hip_bf16.h>

// ---------------- problem constants ----------------
constexpr int N_ = 50000, D_ = 256, H_ = 256, E_ = 800000, G_ = 64, L_ = 3, OUT_ = 2;
constexpr float EPS_ = 1e-5f;
static_assert(H_ == 256 && D_ == 256, "K=256 hardcoded in WMMA loop; shift tricks assume 256");
static_assert((N_ % 16) == 0 && (G_ % 16) == 0, "tile assumptions");

// ---------------- WMMA types ----------------
typedef __attribute__((ext_vector_type(16))) __bf16 v16bf;
typedef __attribute__((ext_vector_type(8)))  float  v8f;

static __device__ __forceinline__ unsigned short f2bf(float f) {
  unsigned int u = __float_as_uint(f);
  unsigned int r = u + 0x7FFFu + ((u >> 16) & 1u);   // round-to-nearest-even
  return (unsigned short)(r >> 16);
}

union BfVec { v16bf v; unsigned short s[16]; uint4 q[2]; };

// Accumulate one 16x16 tile: C += Abf([Mx256] bf16 row-major) * Wt([Noutx256] bf16, row-per-n).
// A fragment (16-bit A 16x32): lane m = lane&15; lanes<16: K 0-7 & 16-23, lanes>=16: +8.
// B fragment (16-bit B 32x16): lane n = lane&15; lanes<16: K 0-7 & 16-23, lanes>=16: +8.
static __device__ __forceinline__ v8f wmma_k256(v8f c,
                                                const unsigned short* __restrict__ Abf,
                                                const unsigned short* __restrict__ Wt,
                                                int m, int n, int koffA, int koffB) {
  const unsigned short* arow = Abf + (size_t)m * 256 + koffA;
  const unsigned short* brow = Wt  + (size_t)n * 256 + koffB;
  __builtin_prefetch(arow + 256, 0, 0);              // global_prefetch_b8: next A row
#pragma unroll
  for (int kb = 0; kb < 256; kb += 32) {
    BfVec au;
    au.q[0] = *(const uint4*)(arow + kb);            // K koffA .. +7   (16B aligned)
    au.q[1] = *(const uint4*)(arow + kb + 16);       // K koffA+16..+23
    BfVec bu;
    bu.q[0] = *(const uint4*)(brow + kb);            // K koffB .. +7   (16B aligned)
    bu.q[1] = *(const uint4*)(brow + kb + 16);       // K koffB+16..+23
    c = __builtin_amdgcn_wmma_f32_16x16x32_bf16(false, au.v, false, bu.v, (short)0, c,
                                                false, false);
  }
  return c;
}

// C[MxNout] = A1@Wt1^T (+ A2@Wt2^T) + bias, optional ReLU. One wave per 16x16 tile. K==256.
__global__ __launch_bounds__(256) void gemm_dual_wmma(
    const unsigned short* __restrict__ A1, const unsigned short* __restrict__ Wt1,
    const unsigned short* __restrict__ A2, const unsigned short* __restrict__ Wt2,
    const float* __restrict__ bias, float* __restrict__ C,
    int M, int Nout, int relu) {
  const int Mt = M >> 4;
  const int tiles = Mt * (Nout >> 4);
  const int wid = blockIdx.x * (blockDim.x >> 5) + (threadIdx.x >> 5);
  if (wid >= tiles) return;                          // wave-uniform: EXEC stays all-ones
  const int lane  = threadIdx.x & 31;
  const int half  = lane >> 4;
  const int mr    = lane & 15;
  const int tm    = wid % Mt;
  const int tn    = wid / Mt;
  const int m     = tm * 16 + mr;
  const int n     = tn * 16 + mr;
  const int koffA = half * 8;
  const int koffB = half * 8;

  v8f c = {};
  c = wmma_k256(c, A1, Wt1, m, n, koffA, koffB);
  if (A2) c = wmma_k256(c, A2, Wt2, m, n, koffA, koffB);

  const float bv = bias ? bias[n] : 0.0f;
  const int mbase = tm * 16 + half * 8;              // D layout: VGPR r -> M = r + 8*half, N = lane&15
#pragma unroll
  for (int r = 0; r < 8; ++r) {
    float v = c[r] + bv;
    if (relu) v = fmaxf(v, 0.0f);
    C[(size_t)(mbase + r) * Nout + n] = v;
  }
}

// ---------------- support kernels ----------------
__global__ void fill_f32(float* p, float v, long long n) {
  long long i = (long long)blockIdx.x * blockDim.x + threadIdx.x;
  if (i < n) p[i] = v;
}

__global__ void cvt_f32_bf16(const float* __restrict__ src, unsigned short* __restrict__ dst,
                             long long n) {
  long long i = (long long)blockIdx.x * blockDim.x + threadIdx.x;
  if (i < n) dst[i] = f2bf(src[i]);
}

// fp32 [KxNout] row-major -> bf16 [NoutxK] (transposed)
__global__ void w_cvt_t(const float* __restrict__ src, unsigned short* __restrict__ dst,
                        int K, int Nout) {
  int idx = blockIdx.x * blockDim.x + threadIdx.x;
  if (idx >= K * Nout) return;
  int k = idx / Nout, n = idx % Nout;
  dst[(size_t)n * K + k] = f2bf(src[idx]);
}

// 64 threads per edge, 4 columns each: agg[dst] += h[src]; deg[dst] += 1
__global__ void edge_scatter(const long long* __restrict__ srcI,
                             const long long* __restrict__ dstI,
                             const float* __restrict__ h, float* __restrict__ agg,
                             float* __restrict__ deg) {
  long long t = (long long)blockIdx.x * blockDim.x + threadIdx.x;
  long long e = t >> 6;
  if (e >= E_) return;
  int lane = (int)(t & 63);
  int s = (int)srcI[e], d = (int)dstI[e];
  float4 v = *(const float4*)(h + (size_t)s * H_ + lane * 4);
  float* ap = agg + (size_t)d * H_ + lane * 4;
  atomicAdd(ap + 0, v.x); atomicAdd(ap + 1, v.y);
  atomicAdd(ap + 2, v.z); atomicAdd(ap + 3, v.w);
  if (lane == 0) atomicAdd(deg + d, 1.0f);
}

// aggbf = bf16( agg / max(deg,1) )   (agg is only consumed by the GEMM, so emit bf16 directly)
__global__ void agg_div_bf(const float* __restrict__ agg, const float* __restrict__ deg,
                           unsigned short* __restrict__ aggbf) {
  long long i = (long long)blockIdx.x * blockDim.x + threadIdx.x;
  if (i >= (long long)N_ * H_) return;
  aggbf[i] = f2bf(agg[i] / fmaxf(deg[i >> 8], 1.0f));
}

// per-column sum / sumsq partial reduction (256 threads == 256 columns)
__global__ void bn_stats(const float* __restrict__ h, float* __restrict__ cs,
                         float* __restrict__ cq, int rows) {
  int c = threadIdx.x;
  float s = 0.f, q = 0.f;
  for (int r = blockIdx.x; r < rows; r += gridDim.x) {
    float v = h[(size_t)r * H_ + c];
    s += v; q += v * v;
  }
  atomicAdd(cs + c, s);
  atomicAdd(cq + c, q);
}

// in place: cs -> scale, cq -> shift
__global__ void bn_finalize(float* cs, float* cq, const float* __restrict__ gamma,
                            const float* __restrict__ beta, int rows) {
  int c = threadIdx.x;
  float inv = 1.0f / (float)rows;
  float mu  = cs[c] * inv;
  float var = cq[c] * inv - mu * mu;
  float sc  = gamma[c] * rsqrtf(var + EPS_);
  cs[c] = sc;
  cq[c] = beta[c] - mu * sc;
}

// h = relu(h*scale+shift) in place (f32, feeds edge gather / pooling) + bf16 shadow for GEMM A
__global__ void bn_relu_apply(float* __restrict__ h, unsigned short* __restrict__ hbf,
                              const float* __restrict__ scale, const float* __restrict__ shift) {
  long long i = (long long)blockIdx.x * blockDim.x + threadIdx.x;
  if (i >= (long long)N_ * H_) return;
  int c = (int)(i & 255);
  float v = fmaxf(h[i] * scale[c] + shift[c], 0.0f);
  h[i] = v;
  hbf[i] = f2bf(v);
}

// 64 threads per node, 4 columns each; values are post-ReLU (>=0) so int atomicMax is exact
__global__ void pool_scatter(const float* __restrict__ h, const long long* __restrict__ batch,
                             float* __restrict__ meanp, float* __restrict__ maxp,
                             float* __restrict__ cnt) {
  long long t = (long long)blockIdx.x * blockDim.x + threadIdx.x;
  long long nidx = t >> 6;
  if (nidx >= N_) return;
  int lane = (int)(t & 63);
  int g = (int)batch[nidx];
  float4 v = *(const float4*)(h + (size_t)nidx * H_ + lane * 4);
  size_t base = (size_t)g * H_ + lane * 4;
  atomicAdd(meanp + base + 0, v.x); atomicAdd(meanp + base + 1, v.y);
  atomicAdd(meanp + base + 2, v.z); atomicAdd(meanp + base + 3, v.w);
  int* mp = (int*)maxp;
  atomicMax(mp + base + 0, __float_as_int(v.x));
  atomicMax(mp + base + 1, __float_as_int(v.y));
  atomicMax(mp + base + 2, __float_as_int(v.z));
  atomicMax(mp + base + 3, __float_as_int(v.w));
  if (lane == 0) atomicAdd(cnt + g, 1.0f);
}

__global__ void pool_div(float* __restrict__ meanp, const float* __restrict__ cnt) {
  int i = blockIdx.x * blockDim.x + threadIdx.x;
  if (i >= G_ * H_) return;
  meanp[i] /= fmaxf(cnt[i >> 8], 1.0f);
}

// out[g,o] = b2[o] + sum_k hid[g,k] * W2[k,o]
__global__ void head_out(const float* __restrict__ hid, const float* __restrict__ W2,
                         const float* __restrict__ b2, float* __restrict__ out) {
  int t = threadIdx.x;
  if (t >= G_ * OUT_) return;
  int g = t / OUT_, o = t % OUT_;
  float acc = b2[o];
  for (int k = 0; k < H_; ++k) acc += hid[(size_t)g * H_ + k] * W2[k * OUT_ + o];
  out[t] = acc;
}

// ---------------- workspace layout ----------------
constexpr size_t alup(size_t x) { return (x + 255) & ~(size_t)255; }
constexpr size_t NHB    = (size_t)N_ * H_ * sizeof(float);
constexpr size_t NHB2   = (size_t)N_ * H_ * 2;        // bf16 shadow
constexpr size_t O_H    = 0;                          // f32 h (single buffer)
constexpr size_t O_HBF  = O_H    + alup(NHB);         // bf16 h
constexpr size_t O_AGG  = O_HBF  + alup(NHB2);        // f32 agg (atomic target)
constexpr size_t O_ABF  = O_AGG  + alup(NHB);         // bf16 agg; also bf16 x before the loop
constexpr size_t O_DEG  = O_ABF  + alup(NHB2);
constexpr size_t O_CS   = O_DEG  + alup((size_t)N_ * 4);
constexpr size_t O_CQ   = O_CS   + alup((size_t)H_ * 4);
constexpr size_t O_WTIN = O_CQ   + alup((size_t)H_ * 4);
constexpr size_t O_WTL  = O_WTIN + alup((size_t)H_ * D_ * 2);
constexpr size_t O_WTR  = O_WTL  + alup((size_t)L_ * H_ * H_ * 2);
constexpr size_t O_W1A  = O_WTR  + alup((size_t)L_ * H_ * H_ * 2);
constexpr size_t O_W1B  = O_W1A  + alup((size_t)H_ * H_ * 2);
constexpr size_t O_MEAN = O_W1B  + alup((size_t)H_ * H_ * 2);
constexpr size_t O_MAX  = O_MEAN + alup((size_t)G_ * H_ * 4);
constexpr size_t O_MEB  = O_MAX  + alup((size_t)G_ * H_ * 4);
constexpr size_t O_MXB  = O_MEB  + alup((size_t)G_ * H_ * 2);
constexpr size_t O_CNT  = O_MXB  + alup((size_t)G_ * H_ * 2);
constexpr size_t O_HID  = O_CNT  + alup((size_t)G_ * 4);

extern "C" void kernel_launch(void* const* d_in, const int* in_sizes, int n_in,
                              void* d_out, int out_size, void* d_ws, size_t ws_size,
                              hipStream_t stream) {
  (void)in_sizes; (void)n_in; (void)out_size; (void)ws_size;

  const float*     x     = (const float*)d_in[0];
  const long long* eidx  = (const long long*)d_in[1];
  const long long* batch = (const long long*)d_in[2];
  const float*     W_in  = (const float*)d_in[3];
  const float*     b_in  = (const float*)d_in[4];
  const float*     W_l   = (const float*)d_in[5];
  const float*     b_l   = (const float*)d_in[6];
  const float*     W_r   = (const float*)d_in[7];
  const float*     gam   = (const float*)d_in[8];
  const float*     bet   = (const float*)d_in[9];
  const float*     W1    = (const float*)d_in[10];
  const float*     b1    = (const float*)d_in[11];
  const float*     W2    = (const float*)d_in[12];
  const float*     b2    = (const float*)d_in[13];
  float* out = (float*)d_out;

  char* ws = (char*)d_ws;
  float*          h     = (float*)(ws + O_H);
  unsigned short* hbf   = (unsigned short*)(ws + O_HBF);
  float*          agg   = (float*)(ws + O_AGG);
  unsigned short* abf   = (unsigned short*)(ws + O_ABF);   // agg-bf16 / x-bf16
  float*          deg   = (float*)(ws + O_DEG);
  float*          cs    = (float*)(ws + O_CS);
  float*          cq    = (float*)(ws + O_CQ);
  unsigned short* wtin  = (unsigned short*)(ws + O_WTIN);
  unsigned short* wtl   = (unsigned short*)(ws + O_WTL);
  unsigned short* wtr   = (unsigned short*)(ws + O_WTR);
  unsigned short* w1a   = (unsigned short*)(ws + O_W1A);
  unsigned short* w1b   = (unsigned short*)(ws + O_W1B);
  float*          meanp = (float*)(ws + O_MEAN);
  float*          maxp  = (float*)(ws + O_MAX);
  unsigned short* mebf  = (unsigned short*)(ws + O_MEB);
  unsigned short* mxbf  = (unsigned short*)(ws + O_MXB);
  float*          cnt   = (float*)(ws + O_CNT);
  float*          hid   = (float*)(ws + O_HID);

  const int TB = 256;
  const long long NH = (long long)N_ * H_;
  const int nhBlocks   = (int)((NH + TB - 1) / TB);
  const int wBlocks    = (H_ * H_ + TB - 1) / TB;
  const int edgeBlocks = (int)(((long long)E_ * 64 + TB - 1) / TB);
  const int poolBlocks = (int)(((long long)N_ * 64 + TB - 1) / TB);
  const int ghBlocks   = (G_ * H_ + TB - 1) / TB;
  const int mainTiles  = (N_ / 16) * (H_ / 16);
  const int mainBlocks = (mainTiles + 7) / 8;          // 8 waves / block (wave32)
  const int headTiles  = (G_ / 16) * (H_ / 16);
  const int headBlocks = (headTiles + 7) / 8;

  // ---- weight convert + transpose to bf16 [Nout x K] ----
  w_cvt_t<<<wBlocks, TB, 0, stream>>>(W_in, wtin, D_, H_);
  for (int i = 0; i < L_; ++i) {
    w_cvt_t<<<wBlocks, TB, 0, stream>>>(W_l + (size_t)i * H_ * H_, wtl + (size_t)i * H_ * H_, H_, H_);
    w_cvt_t<<<wBlocks, TB, 0, stream>>>(W_r + (size_t)i * H_ * H_, wtr + (size_t)i * H_ * H_, H_, H_);
  }
  w_cvt_t<<<wBlocks, TB, 0, stream>>>(W1,                  w1a, H_, H_); // rows 0..H-1   (mean part)
  w_cvt_t<<<wBlocks, TB, 0, stream>>>(W1 + (size_t)H_ * H_, w1b, H_, H_); // rows H..2H-1 (max part)

  // ---- h = x @ W_in + b_in ----
  cvt_f32_bf16<<<nhBlocks, TB, 0, stream>>>(x, abf, (long long)N_ * D_);
  gemm_dual_wmma<<<mainBlocks, TB, 0, stream>>>(abf, wtin, nullptr, nullptr, b_in, h,
                                                N_, H_, 0);
  cvt_f32_bf16<<<nhBlocks, TB, 0, stream>>>(h, hbf, NH);

  const long long* srcI = eidx;        // edge_index[0]
  const long long* dstI = eidx + E_;   // edge_index[1]

  for (int i = 0; i < L_; ++i) {
    // mean-aggregate neighbors (f32 atomics), emit bf16 for the GEMM
    fill_f32<<<nhBlocks, TB, 0, stream>>>(agg, 0.0f, NH);
    fill_f32<<<(N_ + TB - 1) / TB, TB, 0, stream>>>(deg, 0.0f, N_);
    edge_scatter<<<edgeBlocks, TB, 0, stream>>>(srcI, dstI, h, agg, deg);
    agg_div_bf<<<nhBlocks, TB, 0, stream>>>(agg, deg, abf);
    // h = agg @ W_l + b_l + h @ W_r  (dual-accumulate WMMA; reads bf16 shadows, overwrites f32 h)
    gemm_dual_wmma<<<mainBlocks, TB, 0, stream>>>(abf, wtl + (size_t)i * H_ * H_,
                                                  hbf, wtr + (size_t)i * H_ * H_,
                                                  b_l + (size_t)i * H_, h,
                                                  N_, H_, 0);
    // batchnorm + relu (updates f32 h in place and bf16 shadow)
    fill_f32<<<1, TB, 0, stream>>>(cs, 0.0f, H_);
    fill_f32<<<1, TB, 0, stream>>>(cq, 0.0f, H_);
    bn_stats<<<512, TB, 0, stream>>>(h, cs, cq, N_);
    bn_finalize<<<1, H_, 0, stream>>>(cs, cq, gam + (size_t)i * H_, bet + (size_t)i * H_, N_);
    bn_relu_apply<<<nhBlocks, TB, 0, stream>>>(h, hbf, cs, cq);
  }

  // ---- per-graph mean / max pooling ----
  fill_f32<<<ghBlocks, TB, 0, stream>>>(meanp, 0.0f, (long long)G_ * H_);
  fill_f32<<<ghBlocks, TB, 0, stream>>>(maxp, 0.0f, (long long)G_ * H_);
  fill_f32<<<1, TB, 0, stream>>>(cnt, 0.0f, G_);
  pool_scatter<<<poolBlocks, TB, 0, stream>>>(h, batch, meanp, maxp, cnt);
  pool_div<<<ghBlocks, TB, 0, stream>>>(meanp, cnt);
  cvt_f32_bf16<<<ghBlocks, TB, 0, stream>>>(meanp, mebf, (long long)G_ * H_);
  cvt_f32_bf16<<<ghBlocks, TB, 0, stream>>>(maxp, mxbf, (long long)G_ * H_);

  // ---- head: hid = relu([mean|max] @ W1 + b1); out = hid @ W2 + b2 ----
  gemm_dual_wmma<<<headBlocks, TB, 0, stream>>>(mebf, w1a, mxbf, w1b, b1, hid,
                                                G_, H_, 1);
  head_out<<<1, 128, 0, stream>>>(hid, W2, b2, out);
}
